// MultiHeadAttention_80272938762455
// MI455X (gfx1250) — matrix operand order repaired and checked by reading, not compiled, not run on hardware
//
#include <hip/hip_runtime.h>
#include <hip/hip_bf16.h>

// ---------------------------------------------------------------------------
// Distance-kernel (Sinkhorn-style) multi-head attention, MI455X / gfx1250.
// All GEMM work runs on v_wmma_f32_16x16x32_bf16 (wave32 WMMA).
// S x S compat matrices are NEVER materialized in HBM (flash-style fusion).
// Raw bf16 LDS staging uses GLOBAL_LOAD_ASYNC_TO_LDS_B128 (ASYNCcnt) with
// ping-pong double buffering; f32->bf16 converting stages keep the VGPR path.
// ---------------------------------------------------------------------------

typedef unsigned short u16;
typedef unsigned int   u32;

typedef __attribute__((ext_vector_type(16))) __bf16 v16bf;
typedef __attribute__((ext_vector_type(8)))  float  v8f;
typedef __attribute__((ext_vector_type(4)))  int    v4i;

union FragB16 { v16bf v; u32 u[8]; };

#define D_MODEL 1024
#define NUM_H   16
#define DK      64
#define SEQ     2048
#define BATCH   4
#define LDK     72     // bf16 elems per LDS row for K-major tiles (64 + pad, 144B => 16B aligned)
#define LDV     136    // bf16 elems per LDS row for j-major tiles (128 + pad)
#define LDW     136

// ---- gfx1250 async-to-LDS path (guarded; falls back to sync b128 copy) ----
#if defined(__has_builtin)
#  if __has_builtin(__builtin_amdgcn_global_load_async_to_lds_b128) && \
      __has_builtin(__builtin_amdgcn_s_wait_asynccnt)
#    define HAVE_ASYNC_LDS 1
#  endif
#endif
#ifndef HAVE_ASYNC_LDS
#  define HAVE_ASYNC_LDS 0
#endif

__device__ __forceinline__ void async_copy_b128(u16* lds_dst, const u16* gsrc) {
#if HAVE_ASYNC_LDS
  __builtin_amdgcn_global_load_async_to_lds_b128(
      (__attribute__((address_space(1))) v4i*)gsrc,
      (__attribute__((address_space(3))) v4i*)lds_dst, 0, 0);
#else
  *(uint4*)lds_dst = *(const uint4*)gsrc;
#endif
}
__device__ __forceinline__ void async_wait0() {
#if HAVE_ASYNC_LDS
  __builtin_amdgcn_s_wait_asynccnt(0);
#endif
}

__device__ __forceinline__ u16 f2bf(float x) {
  u32 u = __float_as_uint(x);
  u += 0x7FFFu + ((u >> 16) & 1u);
  return (u16)(u >> 16);
}
__device__ __forceinline__ u32 f2bf2(float lo, float hi) {
  return (u32)f2bf(lo) | ((u32)f2bf(hi) << 16);
}
__device__ __forceinline__ float bf2f(u16 h) {
  return __uint_as_float(((u32)h) << 16);
}
__device__ __forceinline__ v8f vzero8() {
  v8f z = {0.f,0.f,0.f,0.f,0.f,0.f,0.f,0.f};
  return z;
}
__device__ __forceinline__ v8f wmma_bf16(v16bf a, v16bf b, v8f c) {
  return __builtin_amdgcn_wmma_f32_16x16x32_bf16(false, a, false, b, (short)0, c, false, false);
}

// A-matrix 16x32 bf16 fragment: lane m = lane&15 holds row m; VGPR v packs
// K = 2v + 8*half (+8 if v>=4). p points at element (row m, k=0).
__device__ __forceinline__ v16bf frag_a(const u16* p, int hf) {
  FragB16 f;
#pragma unroll
  for (int v = 0; v < 8; ++v) {
    int kb = 2 * v + 8 * hf + ((v >= 4) ? 8 : 0);
    f.u[v] = *(const u32*)(p + kb);
  }
  return f.v;
}
// B-matrix 32x16 bf16 fragment: lane n = lane&15 holds column n; VGPR v packs
// K = 2v + 8*half (+8 if v>=4). p points at (col n)'s K-contiguous data (i.e. W row n).
__device__ __forceinline__ v16bf frag_b(const u16* p, int hf) {
  FragB16 f;
#pragma unroll
  for (int v = 0; v < 8; ++v) {
    int kb = 2 * v + 8 * hf + ((v >= 4) ? 8 : 0);
    f.u[v] = *(const u32*)(p + kb);
  }
  return f.v;
}

// ---------------------------------------------------------------------------
// GEMM: C[8192 x 1024] = A[8192 x 1024] * W^T (+ bias).
// mode 0: A = f32, output -> bf16 head-split [B,H,S,DK]
// mode 1: A = bf16 (merged attention, ASYNC-staged), output -> f32 + bias
// Block tile 128x128, K-slab 64 double-buffered, 8 waves (4x2), wave 32x64.
// ---------------------------------------------------------------------------
__global__ __launch_bounds__(256) void gemm_wmma_kernel(
    const float* __restrict__ Af, const u16* __restrict__ Ab,
    const float* __restrict__ W, const float* __restrict__ bias,
    u16* __restrict__ out_bf, float* __restrict__ out_f, int mode) {
  __shared__ u16 As[2][128 * LDK];
  __shared__ u16 Ws[2][128 * LDK];

  const int M0 = blockIdx.x * 128;
  const int N0 = blockIdx.y * 128;
  const int tid = threadIdx.x;
  const int lane = tid & 31, wave = tid >> 5;
  const int hf = lane >> 4, n16 = lane & 15;
  const int wm = wave >> 1, wn = wave & 1;

  v8f acc[2][4];
#pragma unroll
  for (int i = 0; i < 2; ++i)
#pragma unroll
    for (int j = 0; j < 4; ++j) acc[i][j] = vzero8();

  const int rr = tid >> 1;          // 0..127 : staged row
  const int ch = (tid & 1) * 32;    // k-half within slab

  auto stage = [&](int buf, int k0) {
    if (mode == 0) {
      const float* arow = Af + (size_t)(M0 + rr) * D_MODEL + k0 + ch;
#pragma unroll
      for (int i = 0; i < 8; ++i) {
        float4 f = ((const float4*)arow)[i];
        uint2 pk; pk.x = f2bf2(f.x, f.y); pk.y = f2bf2(f.z, f.w);
        *((uint2*)&As[buf][rr * LDK + ch + 4 * i]) = pk;
      }
    } else {
      const u16* asrc = Ab + (size_t)(M0 + rr) * D_MODEL + k0 + ch;
#pragma unroll
      for (int i = 0; i < 4; ++i)
        async_copy_b128(&As[buf][rr * LDK + ch + 8 * i], asrc + 8 * i);
    }
    const float* wrow = W + (size_t)(N0 + rr) * D_MODEL + k0 + ch;
#pragma unroll
    for (int i = 0; i < 8; ++i) {
      float4 f = ((const float4*)wrow)[i];
      uint2 pk; pk.x = f2bf2(f.x, f.y); pk.y = f2bf2(f.z, f.w);
      *((uint2*)&Ws[buf][rr * LDK + ch + 4 * i]) = pk;
    }
  };

  stage(0, 0);
  async_wait0();
  __syncthreads();

  for (int k0 = 0; k0 < D_MODEL; k0 += 64) {
    const int cur = (k0 >> 6) & 1;
    if (k0 + 64 < D_MODEL) {
      stage(cur ^ 1, k0 + 64);                       // in flight while we compute
      __builtin_prefetch(W + (size_t)(N0 + rr) * D_MODEL + k0 + 128 + ch, 0, 1);
    }

#pragma unroll
    for (int ks = 0; ks < 2; ++ks) {
      v16bf af[2], bfg[4];
#pragma unroll
      for (int rt = 0; rt < 2; ++rt)
        af[rt] = frag_a(&As[cur][(32 * wm + 16 * rt + n16) * LDK + 32 * ks], hf);
#pragma unroll
      for (int nt = 0; nt < 4; ++nt)
        bfg[nt] = frag_b(&Ws[cur][(64 * wn + 16 * nt + n16) * LDK + 32 * ks], hf);
#pragma unroll
      for (int rt = 0; rt < 2; ++rt)
#pragma unroll
        for (int nt = 0; nt < 4; ++nt)
          acc[rt][nt] = wmma_bf16(af[rt], bfg[nt], acc[rt][nt]);
    }

    async_wait0();
    __syncthreads();
  }

  // ---- writeback. D layout: lane n = col, VGPR v -> row v + 8*half ----
#pragma unroll
  for (int rt = 0; rt < 2; ++rt) {
#pragma unroll
    for (int nt = 0; nt < 4; ++nt) {
      const int col = N0 + 64 * wn + 16 * nt + n16;
      if (mode == 0) {
        const int h = col >> 6, d = col & 63;
#pragma unroll
        for (int v = 0; v < 8; ++v) {
          const int row = M0 + 32 * wm + 16 * rt + 8 * hf + v;
          const int b = row >> 11, s = row & 2047;
          out_bf[((((size_t)b * NUM_H + h) * SEQ) + s) * DK + d] = f2bf(acc[rt][nt][v]);
        }
      } else {
        const float bi = bias[col];
#pragma unroll
        for (int v = 0; v < 8; ++v) {
          const int row = M0 + 32 * wm + 16 * rt + 8 * hf + v;
          out_f[(size_t)row * D_MODEL + col] = acc[rt][nt][v] + bi;
        }
      }
    }
  }
}

// ---------------------------------------------------------------------------
// Per-head-row squared L2 norms of bf16 Q/K: rows = B*H*S, 64 elems each.
// ---------------------------------------------------------------------------
__global__ __launch_bounds__(256) void rownorm_kernel(const u16* __restrict__ X,
                                                      float* __restrict__ out,
                                                      int nrows) {
  int r = blockIdx.x * 256 + threadIdx.x;
  if (r >= nrows) return;
  const uint4* p = (const uint4*)(X + (size_t)r * DK);
  float s = 0.f;
#pragma unroll
  for (int i = 0; i < 8; ++i) {
    uint4 w = p[i];
    u32 a[4] = {w.x, w.y, w.z, w.w};
#pragma unroll
    for (int j = 0; j < 4; ++j) {
      float lo = bf2f((u16)a[j]);
      float hi = bf2f((u16)(a[j] >> 16));
      s += lo * lo + hi * hi;
    }
  }
  out[r] = s;
}

// power-law compat: (1 + dist/64)^(-65)
__device__ __forceinline__ float compat_fn(float q2, float k2, float qk) {
  float d2 = q2 + k2 - 2.f * qk;
  float g = sqrtf(fmaxf(d2, 0.f)) * (1.f / 64.f);
  return __powf(1.f + g, -65.f);
}

// ---------------------------------------------------------------------------
// Column sums of compat -> ncw[j] = N_C[j]^{-1/2}.  grid (S/128, B*H).
// Each wave owns one 16-col tile; 128-row chunks ASYNC double-buffered.
// ---------------------------------------------------------------------------
__global__ __launch_bounds__(256) void colsum_kernel(
    const u16* __restrict__ Qb, const u16* __restrict__ Kb,
    const float* __restrict__ q2, const float* __restrict__ k2,
    float* __restrict__ ncw) {
  __shared__ u16 Qsh[2][128 * LDK];
  __shared__ float q2sh[2][128];

  const int bh = blockIdx.y;
  const int J0 = blockIdx.x * 128;
  const int tid = threadIdx.x;
  const int lane = tid & 31, wave = tid >> 5;
  const int hf = lane >> 4, n16 = lane & 15;

  // B fragments: k-rows j = J0 + 16*wave + n16, K = d (contig 64 bf16). Load once.
  const u16* krow = Kb + ((size_t)bh * SEQ + J0 + 16 * wave + n16) * DK;
  const v16bf bk0 = frag_b(krow, hf);
  const v16bf bk1 = frag_b(krow + 32, hf);
  const float k2v = k2[(size_t)bh * SEQ + J0 + 16 * wave + n16];

  float csum = 0.f;
  const u16* qbase = Qb + (size_t)bh * SEQ * DK;

  auto stage = [&](int buf, int i0) {
    const int jl = tid >> 1, off = (tid & 1) * 32;
    const u16* src = qbase + (size_t)(i0 + jl) * DK + off;
#pragma unroll
    for (int i = 0; i < 4; ++i)
      async_copy_b128(&Qsh[buf][jl * LDK + off + 8 * i], src + 8 * i);
    if (tid < 128) q2sh[buf][tid] = q2[(size_t)bh * SEQ + i0 + tid];
  };

  stage(0, 0);
  async_wait0();
  __syncthreads();

  for (int i0 = 0; i0 < SEQ; i0 += 128) {
    const int cur = (i0 >> 7) & 1;
    if (i0 + 128 < SEQ) stage(cur ^ 1, i0 + 128);

#pragma unroll
    for (int rt = 0; rt < 8; ++rt) {
      v8f c = vzero8();
      const u16* ar = &Qsh[cur][(16 * rt + n16) * LDK];
      c = wmma_bf16(frag_a(ar, hf), bk0, c);
      c = wmma_bf16(frag_a(ar + 32, hf), bk1, c);
#pragma unroll
      for (int v = 0; v < 8; ++v) {
        const int rl = 16 * rt + 8 * hf + v;
        csum += compat_fn(q2sh[cur][rl], k2v, c[v]);
      }
    }

    async_wait0();
    __syncthreads();
  }
  // combine the two half-wave contributions for each column
  csum += __shfl_xor(csum, 16, 32);
  if (lane < 16)
    ncw[(size_t)bh * SEQ + J0 + 16 * wave + lane] = rsqrtf(csum);
}

// ---------------------------------------------------------------------------
// Fused attention pass.  grid (S/128, B*H); wave w owns rows 16w..16w+15.
// Per 128-col K/V block (ASYNC double-buffered): WMMA qk -> compat*ncw ->
// bf16 strip in LDS -> WMMA vs transposed-V (+ ones tile => row sums).
// ---------------------------------------------------------------------------
__global__ __launch_bounds__(256) void attn_kernel(
    const u16* __restrict__ Qb, const u16* __restrict__ Kb,
    const u16* __restrict__ Vb, const float* __restrict__ q2,
    const float* __restrict__ k2, const float* __restrict__ ncw,
    u16* __restrict__ merged) {
  __shared__ u16 Ks[2][128 * LDK];      // k rows, d-major
  __shared__ u16 Vt[2][DK * LDV];       // V transposed: [d][j]
  __shared__ u16 Wst[8 * 16 * LDW];     // per-wave bf16 weight strips
  __shared__ float k2s[2][128];
  __shared__ float ncws[2][128];

  const int bh = blockIdx.y, b = bh >> 4, h = bh & 15;
  const int I0 = blockIdx.x * 128;
  const int tid = threadIdx.x;
  const int lane = tid & 31, wave = tid >> 5;
  const int hf = lane >> 4, n16 = lane & 15;

  // Q A-fragments for this wave's 16 rows (direct per-lane global gather, once).
  const u16* qrow = Qb + ((size_t)bh * SEQ + I0 + 16 * wave + n16) * DK;
  const v16bf aq0 = frag_a(qrow, hf);
  const v16bf aq1 = frag_a(qrow + 32, hf);

  float q2r[8];
#pragma unroll
  for (int v = 0; v < 8; ++v)
    q2r[v] = q2[(size_t)bh * SEQ + I0 + 16 * wave + 8 * hf + v];

  FragB16 onesf;  // B tile with column 0 == 1.0 (row-sum trick)
#pragma unroll
  for (int v = 0; v < 8; ++v) onesf.u[v] = (n16 == 0) ? 0x3F803F80u : 0u;

  v8f acc[5];
#pragma unroll
  for (int t = 0; t < 5; ++t) acc[t] = vzero8();

  u16* wst = &Wst[wave * 16 * LDW];

  auto stage = [&](int buf, int J0) {
    const int jl = tid >> 1, off = (tid & 1) * 32;
    // K block: raw bf16 copy -> async to LDS
    const u16* ksrc = Kb + ((size_t)bh * SEQ + J0 + jl) * DK + off;
#pragma unroll
    for (int i = 0; i < 4; ++i)
      async_copy_b128(&Ks[buf][jl * LDK + off + 8 * i], ksrc + 8 * i);
    // V block transposed -> Vt[d][j] (needs VGPR path)
    const uint2* vsrc = (const uint2*)(Vb + ((size_t)bh * SEQ + J0 + jl) * DK + off);
#pragma unroll
    for (int i = 0; i < 8; ++i) {
      uint2 w = vsrc[i];
      const int d0 = off + 4 * i;
      Vt[buf][(d0 + 0) * LDV + jl] = (u16)(w.x);
      Vt[buf][(d0 + 1) * LDV + jl] = (u16)(w.x >> 16);
      Vt[buf][(d0 + 2) * LDV + jl] = (u16)(w.y);
      Vt[buf][(d0 + 3) * LDV + jl] = (u16)(w.y >> 16);
    }
    if (tid < 128) k2s[buf][tid] = k2[(size_t)bh * SEQ + J0 + tid];
    else           ncws[buf][tid - 128] = ncw[(size_t)bh * SEQ + J0 + (tid - 128)];
  };

  stage(0, 0);
  async_wait0();
  __syncthreads();

  for (int J0 = 0; J0 < SEQ; J0 += 128) {
    const int cur = (J0 >> 7) & 1;
    if (J0 + 128 < SEQ) stage(cur ^ 1, J0 + 128);

    // --- compat row-strip: 16 rows x 128 cols per wave ---
#pragma unroll
    for (int ct = 0; ct < 8; ++ct) {
      v8f c = vzero8();
      const u16* kr = &Ks[cur][(16 * ct + n16) * LDK];
      c = wmma_bf16(aq0, frag_b(kr, hf), c);
      c = wmma_bf16(aq1, frag_b(kr + 32, hf), c);
      const int cl = 16 * ct + n16;
      const float k2v = k2s[cur][cl];
      const float nw = ncws[cur][cl];
#pragma unroll
      for (int v = 0; v < 8; ++v) {
        float wgt = compat_fn(q2r[v], k2v, c[v]) * nw;
        wst[(v + 8 * hf) * LDW + cl] = f2bf(wgt);
      }
    }
    // same-wave LDS RAW (DS pipe is in-order per wave; compiler inserts waits)

    // --- weights x [V | 1]: accumulate O (4 tiles) + row sums (tile 4) ---
#pragma unroll
    for (int ks = 0; ks < 4; ++ks) {
      const v16bf aw = frag_a(&wst[n16 * LDW + 32 * ks], hf);
#pragma unroll
      for (int dt = 0; dt < 4; ++dt) {
        const v16bf bv = frag_b(&Vt[cur][(16 * dt + n16) * LDV + 32 * ks], hf);
        acc[dt] = wmma_bf16(aw, bv, acc[dt]);
      }
      acc[4] = wmma_bf16(aw, onesf.v, acc[4]);
    }

    async_wait0();
    __syncthreads();
  }

  // --- normalize rows and store merged [B, S, H*DK] ---
#pragma unroll
  for (int v = 0; v < 8; ++v) {
    const float rs = __shfl(acc[4][v], lane & 16, 32);  // row-sum lives in lane 0/16
    const float inv = 1.f / fmaxf(rs, 1e-12f);
    const int s = I0 + 16 * wave + 8 * hf + v;
    const size_t base = ((size_t)b * SEQ + s) * D_MODEL + h * DK;
#pragma unroll
    for (int dt = 0; dt < 4; ++dt)
      merged[base + dt * 16 + n16] = f2bf(acc[dt][v] * inv);
  }
}

// ---------------------------------------------------------------------------
extern "C" void kernel_launch(void* const* d_in, const int* in_sizes, int n_in,
                              void* d_out, int out_size, void* d_ws, size_t ws_size,
                              hipStream_t stream) {
  const float* queries = (const float*)d_in[0];
  const float* keys    = (const float*)d_in[1];
  const float* values  = (const float*)d_in[2];
  const float* Wq = (const float*)d_in[3];
  const float* Wk = (const float*)d_in[4];
  const float* Wv = (const float*)d_in[5];
  const float* Wo = (const float*)d_in[6];
  const float* bo = (const float*)d_in[7];
  float* out = (float*)d_out;

  const size_t nBHS = (size_t)BATCH * NUM_H * SEQ;          // 131072 head-rows
  const size_t qkvBytes = nBHS * DK * sizeof(u16);          // 16 MB each
  char* ws = (char*)d_ws;
  size_t off = 0;
  auto alloc = [&](size_t bytes) {
    void* p = ws + off;
    off = (off + bytes + 255) & ~(size_t)255;
    return p;
  };
  u16* Qb = (u16*)alloc(qkvBytes);
  u16* Kb = (u16*)alloc(qkvBytes);
  u16* Vb = (u16*)alloc(qkvBytes);
  u16* merged = (u16*)alloc((size_t)BATCH * SEQ * D_MODEL * sizeof(u16));
  float* q2  = (float*)alloc(nBHS * sizeof(float));
  float* k2  = (float*)alloc(nBHS * sizeof(float));
  float* ncw = (float*)alloc(nBHS * sizeof(float));
  (void)ws_size; (void)in_sizes; (void)n_in; (void)out_size;

  dim3 gemmGrid(64, 8);   // 8192/128 x 1024/128
  // 1) projections -> head-split bf16
  gemm_wmma_kernel<<<gemmGrid, 256, 0, stream>>>(queries, nullptr, Wq, nullptr, Qb, nullptr, 0);
  gemm_wmma_kernel<<<gemmGrid, 256, 0, stream>>>(keys,    nullptr, Wk, nullptr, Kb, nullptr, 0);
  gemm_wmma_kernel<<<gemmGrid, 256, 0, stream>>>(values,  nullptr, Wv, nullptr, Vb, nullptr, 0);
  // 2) squared norms (from bf16-rounded data, consistent with WMMA qk)
  rownorm_kernel<<<(int)(nBHS / 256), 256, 0, stream>>>(Qb, q2, (int)nBHS);
  rownorm_kernel<<<(int)(nBHS / 256), 256, 0, stream>>>(Kb, k2, (int)nBHS);
  // 3) Sinkhorn column sums -> N_C^{-1/2}
  colsum_kernel<<<dim3(SEQ / 128, BATCH * NUM_H), 256, 0, stream>>>(Qb, Kb, q2, k2, ncw);
  // 4) fused attention -> merged bf16 [B, S, D_MODEL]
  attn_kernel<<<dim3(SEQ / 128, BATCH * NUM_H), 256, 0, stream>>>(Qb, Kb, Vb, q2, k2, ncw, merged);
  // 5) output projection + bias -> f32
  gemm_wmma_kernel<<<gemmGrid, 256, 0, stream>>>(nullptr, merged, Wo, bo, nullptr, out, 1);
}
